// Gumbel_Token_Selection_Block_Wrapper_24180665877021
// MI455X (gfx1250) — matrix operand order repaired and checked by reading, not compiled, not run on hardware
//
#include <hip/hip_runtime.h>
#include <hip/hip_bf16.h>

// ---------------------------------------------------------------------------
// Types for CDNA5 WMMA (wave32): 16x16x32 bf16 -> f32
// ---------------------------------------------------------------------------
typedef __attribute__((ext_vector_type(16))) __bf16 v16bf;
typedef __attribute__((ext_vector_type(8)))  float  v8f;

#define B_DIM    64
#define N_TOK    577
#define P_TOK    576
#define D_DIM    1024
#define K_ALPHA  144
#define OUT_ROWS 145   // 1 cls + 144 kept
#define LDA      1032  // padded LDS row stride (bf16 elems): 516 dwords = 4-bank skew

// workspace layout in floats:
//   [0,      65536)  m_cls        (64 x 1024)
//   [65536, 102400)  y_raw        (64 x 576)
//   [102400,102464)  entH per-b   (64)
//   [102464,111680)  st           (64 x 144)
//   [111680,120896)  idx (int)    (64 x 144)
//   [120896,120896+524288) w_tri bf16 (1024 x 1024, 2 MB)

// ---------------------------------------------------------------------------
// Kernel 0: one-shot f32 -> bf16 conversion of w_tri (stays hot in 192MB L2)
// ---------------------------------------------------------------------------
__global__ __launch_bounds__(256)
void wcvt_kernel(const float* __restrict__ w, __bf16* __restrict__ wbf)
{
    const size_t i = ((size_t)blockIdx.x * 256 + threadIdx.x) * 4;
    float4 q = *reinterpret_cast<const float4*>(w + i);
    __bf16* o = wbf + i;
    o[0] = (__bf16)q.x; o[1] = (__bf16)q.y; o[2] = (__bf16)q.z; o[3] = (__bf16)q.w;
}

// ---------------------------------------------------------------------------
// Kernel 1: m_cls[b,e] = dot(x[b,0,:], w_u_w[e,:]) + w_u_b[e]
// ---------------------------------------------------------------------------
__global__ __launch_bounds__(256)
void mcls_kernel(const float* __restrict__ x, const float* __restrict__ w,
                 const float* __restrict__ bias, float* __restrict__ mcls)
{
    __shared__ float sx[D_DIM];
    const int b = blockIdx.x;
    const int eblk = blockIdx.y;
    const float* xrow = x + (size_t)b * N_TOK * D_DIM;   // cls token row
    for (int i = threadIdx.x; i < D_DIM; i += 256) sx[i] = xrow[i];
    __syncthreads();

    const int e = eblk * 256 + threadIdx.x;
    const float* wr = w + (size_t)e * D_DIM;
    float acc = 0.0f;
#pragma unroll 4
    for (int k = 0; k < D_DIM; k += 4) {
        float4 q = *reinterpret_cast<const float4*>(wr + k);
        acc += sx[k] * q.x + sx[k + 1] * q.y + sx[k + 2] * q.z + sx[k + 3] * q.w;
    }
    mcls[b * D_DIM + e] = acc + bias[e];
}

// ---------------------------------------------------------------------------
// Kernel 2: fused  p_tri = patch @ w_tri^T + b_tri  (bf16 WMMA)
//           y_raw[b,p] = sqrt( sum_e ((p_tri[b,p,e]) * m_cls[b,e])^2 )
// One workgroup = one batch b, one tile of 32 patch rows (2 x 16-row WMMA
// sub-tiles). 8 waves; each wave owns 8 e-tiles of 16 columns, accumulators
// held in registers across the whole K loop (16 x v8f = 128 VGPRs).
// Loop nest: kb outer (no unroll), e-tiles inner (full unroll) so the two
// A fragments loaded per kb from LDS feed all 16 WMMAs of that step.
// ---------------------------------------------------------------------------
__global__ __launch_bounds__(256)
void ptri_ysq_kernel(const float* __restrict__ x, const __bf16* __restrict__ wbf,
                     const float* __restrict__ wtri_b, const float* __restrict__ mcls,
                     float* __restrict__ y_raw)
{
    __shared__ __bf16 ldsA[32 * LDA];   // 32 rows x 1024 k (padded), ~66 KB
    __shared__ float  s_ysq[32];

    const int bid   = blockIdx.x;
    const int b     = bid / (P_TOK / 32);
    const int ptile = bid % (P_TOK / 32);
    const int tid   = threadIdx.x;

    // stage A tile: rows (1 + 32*ptile .. +31) of x[b], f32 -> bf16, skewed rows
    const float* asrc = x + ((size_t)b * N_TOK + 1 + (size_t)ptile * 32) * D_DIM;
    for (int i = tid; i < 32 * D_DIM; i += 256) {
        const int row = i >> 10, col = i & (D_DIM - 1);
        ldsA[row * LDA + col] = (__bf16)asrc[i];
    }
    if (tid < 32) s_ysq[tid] = 0.0f;
    __syncthreads();

    const int wave = tid >> 5;
    const int lane = tid & 31;
    const int m    = lane & 15;   // row-in-subtile for A, col-in-tile for B/C
    const int h    = lane >> 4;   // lane-half -> K sub-chunk select (ISA 7.12.2)

    // 16 accumulator tiles: e-tiles (wave + 8t), sub-tiles 0/1
    v8f c0[8], c1[8];
#pragma unroll
    for (int t = 0; t < 8; ++t) { c0[t] = (v8f){}; c1[t] = (v8f){}; }

    // B column base for this lane: row (e_base + m) of wbf, K chunks {h*8..} and {16+h*8..}
    const __bf16* bcol = wbf + (size_t)m * D_DIM + h * 8;

#pragma unroll 1
    for (int kb = 0; kb < D_DIM; kb += 32) {
        // --- A fragments (16x32 bf16): lane half h -> chunks {h*8..} and {16+h*8..}
        union { uint4 u[2]; v16bf v; } a0, a1;
        const uint4* ap0 =
            reinterpret_cast<const uint4*>(&ldsA[m * LDA + kb + h * 8]);
        const uint4* ap1 =
            reinterpret_cast<const uint4*>(&ldsA[(16 + m) * LDA + kb + h * 8]);
        a0.u[0] = ap0[0]; a0.u[1] = ap0[2];   // K: kb+h*8..+7 , kb+16+h*8..+7
        a1.u[0] = ap1[0]; a1.u[1] = ap1[2];

#pragma unroll
        for (int t = 0; t < 8; ++t) {
            const int e_base = (wave + t * 8) * 16;
            // --- B fragment (32x16 bf16): lane n=m holds K kb+h*8..+7 , kb+16+h*8..+7
            union { uint4 u[2]; v16bf v; } bu;
            const uint4* bp =
                reinterpret_cast<const uint4*>(bcol + (size_t)e_base * D_DIM + kb);
            __builtin_prefetch(bcol + (size_t)e_base * D_DIM + kb + 32, 0, 3);
            bu.u[0] = bp[0]; bu.u[1] = bp[2];

            // D = A x B + C   (2x v_wmma_f32_16x16x32_bf16, A+B reused)
            c0[t] = __builtin_amdgcn_wmma_f32_16x16x32_bf16(
                        false, a0.v, false, bu.v, (short)0, c0[t], false, false);
            c1[t] = __builtin_amdgcn_wmma_f32_16x16x32_bf16(
                        false, a1.v, false, bu.v, (short)0, c1[t], false, false);
        }
    }

    // epilogue: this lane's column e; C VGPR i holds rows i (lanes 0-15) / i+8
    float ysq0[8], ysq1[8];
#pragma unroll
    for (int i = 0; i < 8; ++i) { ysq0[i] = 0.0f; ysq1[i] = 0.0f; }
#pragma unroll
    for (int t = 0; t < 8; ++t) {
        const int e      = (wave + t * 8) * 16 + m;
        const float mc   = mcls[b * D_DIM + e];
        const float bias = wtri_b[e];
#pragma unroll
        for (int i = 0; i < 8; ++i) {
            float v0 = (c0[t][i] + bias) * mc;
            float v1 = (c1[t][i] + bias) * mc;
            ysq0[i] += v0 * v0;
            ysq1[i] += v1 * v1;
        }
    }
    // rows this lane contributes to: i + 8*h (+16) — reduce via ds_add_f32
#pragma unroll
    for (int i = 0; i < 8; ++i) {
        atomicAdd(&s_ysq[i + 8 * h], ysq0[i]);
        atomicAdd(&s_ysq[16 + i + 8 * h], ysq1[i]);
    }
    __syncthreads();
    if (tid < 32)
        y_raw[b * P_TOK + ptile * 32 + tid] = sqrtf(s_ysq[tid]);
}

// ---------------------------------------------------------------------------
// block reductions (wave32)
// ---------------------------------------------------------------------------
__device__ __forceinline__ float blockReduceSum(float v, float* sred)
{
#pragma unroll
    for (int off = 16; off > 0; off >>= 1) v += __shfl_down(v, off, 32);
    const int wave = threadIdx.x >> 5, lane = threadIdx.x & 31;
    if (lane == 0) sred[wave] = v;
    __syncthreads();
    if (threadIdx.x == 0) {
        float t = 0.0f;
        const int nw = (blockDim.x + 31) >> 5;
        for (int i = 0; i < nw; ++i) t += sred[i];
        sred[0] = t;
    }
    __syncthreads();
    float r = sred[0];
    __syncthreads();
    return r;
}

__device__ __forceinline__ float blockReduceMax(float v, float* sred)
{
#pragma unroll
    for (int off = 16; off > 0; off >>= 1) v = fmaxf(v, __shfl_down(v, off, 32));
    const int wave = threadIdx.x >> 5, lane = threadIdx.x & 31;
    if (lane == 0) sred[wave] = v;
    __syncthreads();
    if (threadIdx.x == 0) {
        float t = -3.4e38f;
        const int nw = (blockDim.x + 31) >> 5;
        for (int i = 0; i < nw; ++i) t = fmaxf(t, sred[i]);
        sred[0] = t;
    }
    __syncthreads();
    float r = sred[0];
    __syncthreads();
    return r;
}

// ---------------------------------------------------------------------------
// Kernel 3: per-batch LayerNorm + logits + softmax + entropy + top-144 rank
// ---------------------------------------------------------------------------
__global__ __launch_bounds__(P_TOK)
void head_kernel(const float* __restrict__ y_raw, const float* __restrict__ cls_attn,
                 const float* __restrict__ ema, const float* __restrict__ beta,
                 const float* __restrict__ gammap, const float* __restrict__ gate,
                 const float* __restrict__ bn_w, const float* __restrict__ bn_b,
                 int* __restrict__ idx_ws, float* __restrict__ st_ws,
                 float* __restrict__ entH)
{
    __shared__ float s_probs[P_TOK];
    __shared__ float s_red[32];
    const int b = blockIdx.x;
    const int p = threadIdx.x;

    const float y   = y_raw[b * P_TOK + p];
    const float mu  = blockReduceSum(y, s_red) * (1.0f / P_TOK);
    const float ex2 = blockReduceSum(y * y, s_red) * (1.0f / P_TOK);
    const float var = ex2 - mu * mu;
    const float ytri = (y - mu) * rsqrtf(var + 1e-5f) * bn_w[0] + bn_b[0];

    const float ev    = ema[p];
    const float emam  = blockReduceSum(ev, s_red) * (1.0f / P_TOK);
    const float g     = 1.0f / (1.0f + expf(-gate[0]));
    const float logit = beta[0] * cls_attn[b * N_TOK + 1 + p]
                      + g * gammap[0] * ytri
                      + 0.3f * (ev - emam);

    const float mx   = blockReduceMax(logit, s_red);
    const float e    = expf(logit - mx);
    const float ssum = blockReduceSum(e, s_red);
    const float prob = e / ssum;
    s_probs[p] = prob;

    const float hsum = blockReduceSum(-prob * logf(prob + 1e-9f), s_red);
    if (p == 0) entH[b] = hsum;
    __syncthreads();

    // exact top_k ordering: rank = #{j : probs[j] > probs[p] or (== and j < p)}
    int cnt = 0;
    for (int j = 0; j < P_TOK; ++j) {
        const float q = s_probs[j];
        cnt += (q > prob) || (q == prob && j < p);
    }
    if (cnt < K_ALPHA) {
        idx_ws[b * K_ALPHA + cnt] = p;
        st_ws[b * K_ALPHA + cnt]  = (1.0f - prob) + prob;   // hard - sg(p) + p
    }
}

// ---------------------------------------------------------------------------
// Kernel 4: out[b,0,:] = cls ; out[b,1+r,:] = patch[idx]*st
// ---------------------------------------------------------------------------
__global__ __launch_bounds__(256)
void gather_kernel(const float* __restrict__ x, const int* __restrict__ idx_ws,
                   const float* __restrict__ st_ws, float* __restrict__ out)
{
    const int rb = blockIdx.x;
    const int b  = rb / OUT_ROWS;
    const int r  = rb % OUT_ROWS;
    const float* src;
    float scale;
    if (r == 0) {
        src = x + (size_t)b * N_TOK * D_DIM;
        scale = 1.0f;
    } else {
        const int pidx = idx_ws[b * K_ALPHA + (r - 1)];
        scale = st_ws[b * K_ALPHA + (r - 1)];
        src = x + ((size_t)b * N_TOK + 1 + pidx) * D_DIM;
    }
    float4 q = reinterpret_cast<const float4*>(src)[threadIdx.x];
    q.x *= scale; q.y *= scale; q.z *= scale; q.w *= scale;
    reinterpret_cast<float4*>(out + ((size_t)b * OUT_ROWS + r) * D_DIM)[threadIdx.x] = q;
}

// ---------------------------------------------------------------------------
// Kernel 5: ent_loss = ENT_W * max(mean_b H_b - H_TARGET, 0)
// ---------------------------------------------------------------------------
__global__ void finalize_kernel(const float* __restrict__ entH, float* __restrict__ out_ent)
{
    if (threadIdx.x == 0) {
        float s = 0.0f;
        for (int i = 0; i < B_DIM; ++i) s += entH[i];
        const float H = s * (1.0f / B_DIM);
        out_ent[0] = 0.05f * fmaxf(H - 2.0f, 0.0f);
    }
}

// ---------------------------------------------------------------------------
extern "C" void kernel_launch(void* const* d_in, const int* in_sizes, int n_in,
                              void* d_out, int out_size, void* d_ws, size_t ws_size,
                              hipStream_t stream)
{
    const float* x        = (const float*)d_in[0];
    const float* cls_attn = (const float*)d_in[1];
    const float* w_u_w    = (const float*)d_in[2];
    const float* w_u_b    = (const float*)d_in[3];
    const float* w_tri_w  = (const float*)d_in[4];
    const float* w_tri_b  = (const float*)d_in[5];
    const float* beta     = (const float*)d_in[6];
    const float* gammap   = (const float*)d_in[7];
    const float* gate     = (const float*)d_in[8];
    const float* bn_w     = (const float*)d_in[9];
    const float* bn_b     = (const float*)d_in[10];
    const float* ema      = (const float*)d_in[11];
    (void)in_sizes; (void)n_in; (void)out_size; (void)ws_size;

    float*  ws    = (float*)d_ws;
    float*  mcls  = ws;               // 65536
    float*  y_raw = ws + 65536;       // 36864
    float*  entH  = ws + 102400;      // 64
    float*  st    = ws + 102464;      // 9216
    int*    idx   = (int*)(ws + 111680);   // 9216 ints
    __bf16* wbf   = (__bf16*)(ws + 120896); // 1024x1024 bf16 (2 MB)
    float*  out   = (float*)d_out;

    wcvt_kernel<<<dim3((D_DIM * D_DIM) / (256 * 4)), 256, 0, stream>>>(w_tri_w, wbf);
    mcls_kernel<<<dim3(B_DIM, D_DIM / 256), 256, 0, stream>>>(x, w_u_w, w_u_b, mcls);
    ptri_ysq_kernel<<<dim3(B_DIM * (P_TOK / 32)), 256, 0, stream>>>(
        x, wbf, w_tri_b, mcls, y_raw);
    head_kernel<<<dim3(B_DIM), P_TOK, 0, stream>>>(
        y_raw, cls_attn, ema, beta, gammap, gate, bn_w, bn_b, idx, st, entH);
    gather_kernel<<<dim3(B_DIM * OUT_ROWS), 256, 0, stream>>>(x, idx, st, out);
    finalize_kernel<<<1, 32, 0, stream>>>(entH, out + (size_t)B_DIM * OUT_ROWS * D_DIM);
}
